// Attention_48000554500172
// MI455X (gfx1250) — compile-verified, matrix operand order checked by reading, not run on hardware
//
#include <hip/hip_runtime.h>
#include <hip/hip_bf16.h>

// MI455X (gfx1250) causal MHA + RoPE.
// All matmuls on v_wmma_f32_16x16x32_bf16 (fp32 accum); softmax/RoPE in fp32 VALU.
// Double-buffered LDS pipelines; K/V staged via global_load_async_to_lds_b128
// when the toolchain exposes it (ASYNCcnt + s_wait_asynccnt), else sync copies.
// S=4096, D=2048, H=16, DH=128.

#define S_LEN  4096
#define DMODEL 2048
#define NHEAD  16
#define DHEAD  128

typedef __bf16        v16bf __attribute__((ext_vector_type(16)));
typedef float         v8f   __attribute__((ext_vector_type(8)));
typedef unsigned int  u32x4 __attribute__((ext_vector_type(4)));
typedef unsigned int  u32x2 __attribute__((ext_vector_type(2)));
typedef float         f32x4 __attribute__((ext_vector_type(4)));
typedef int           i32x4 __attribute__((ext_vector_type(4)));

// pointer-to-int4 in global (AS1) / LDS (AS3) address spaces for the async builtin
typedef __attribute__((address_space(1))) i32x4* gas_b128_t;
typedef __attribute__((address_space(3))) i32x4* las_b128_t;

#if defined(__has_builtin)
#if __has_builtin(__builtin_amdgcn_global_load_async_to_lds_b128) && \
    __has_builtin(__builtin_amdgcn_s_wait_asynccnt)
#define USE_ASYNC_LDS 1
#endif
#endif
#ifndef USE_ASYNC_LDS
#define USE_ASYNC_LDS 0
#endif

__device__ __forceinline__ unsigned short f2bf(float f) {
  unsigned int u = __float_as_uint(f);
  u += 0x7FFFu + ((u >> 16) & 1u);          // round-to-nearest-even
  return (unsigned short)(u >> 16);
}

union FragU { v16bf v; u32x4 q[2]; };

// bf16 WMMA operand fragment: lane reads 8 halfwords at rowbase[col..col+7] and
// rowbase[col+16..col+23]; with col = (lane>=16 ? 8 : 0) this matches the 16-bit
// 16x32 operand layout (K 0-7/16-23 in lanes 0-15, K 8-15/24-31 in lanes 16-31).
__device__ __forceinline__ v16bf ld_frag(const unsigned short* rowbase, int col) {
  FragU f;
  f.q[0] = *reinterpret_cast<const u32x4*>(rowbase + col);
  f.q[1] = *reinterpret_cast<const u32x4*>(rowbase + col + 16);
  return f.v;
}

__device__ __forceinline__ v8f wmma_bf16(v16bf a, v16bf b, v8f c) {
  return __builtin_amdgcn_wmma_f32_16x16x32_bf16(false, a, false, b, (short)0, c,
                                                 false, false);
}

// 16-byte global -> LDS stage: async (ASYNCcnt) when available, else via VGPRs.
__device__ __forceinline__ void stage16(unsigned short* lds_dst,
                                        const unsigned short* gsrc) {
#if USE_ASYNC_LDS
  __builtin_amdgcn_global_load_async_to_lds_b128(
      (gas_b128_t)(unsigned long long)gsrc,
      (las_b128_t)(unsigned int)(unsigned long long)lds_dst,
      0, 0);
#else
  *reinterpret_cast<u32x4*>(lds_dst) = *reinterpret_cast<const u32x4*>(gsrc);
#endif
}

__device__ __forceinline__ void wait_stage() {
#if USE_ASYNC_LDS
  __builtin_amdgcn_s_wait_asynccnt(0);
#endif
}

// ---------------------------------------------------------------------------
// GEMM: C[r0+128, n0+128] = A[S, 2048] * W[N, 2048]^T (dot along contiguous K).
// A fp32 (ABF16=0) or bf16 halfwords (ABF16=1); W always fp32 (converted in regs).
// OUTMODE 0: bf16 [H][S][DH] (+optional RoPE)   (Q, K projections)
// OUTMODE 1: bf16 [H][DH][S] transposed         (V projection, for async staging)
// OUTMODE 2: fp32 [S][2048]                     (final output projection)
// Double-buffered LDS; next tile's global loads issued before current compute.
// ---------------------------------------------------------------------------
template<bool ABF16, bool ROPE, int OUTMODE>
__global__ __launch_bounds__(256) void gemm_nt_kernel(
    const void* __restrict__ Ap,
    const float* __restrict__ W,
    void* __restrict__ Out)
{
  __shared__ unsigned short As[2][128 * 40];  // 128 rows x 32 K, padded stride 40
  __shared__ unsigned short Bs[2][128 * 40];

  const int tid  = threadIdx.x;
  const int lane = tid & 31;
  const int w    = tid >> 5;
  const int lr   = lane & 15;
  const int half = lane >> 4;
  const int wr   = w & 3;        // wave row (32-row slab)
  const int wc   = w >> 2;       // wave col (64-col slab)
  const int r0   = blockIdx.y * 128;
  const int n0   = blockIdx.x * 128;

  const float*          A32 = (const float*)Ap;
  const unsigned short* A16 = (const unsigned short*)Ap;

  f32x4 ra[4];
  u32x2 ra16[4];
  f32x4 rw[4];

  auto load_tiles = [&](int k0) {
#pragma unroll
    for (int t = 0; t < 4; ++t) {
      const int chunk = tid + t * 256;      // 1024 4-element chunks
      const int r = chunk >> 3;             // 0..127
      const int c = (chunk & 7) << 2;       // 0,4,..,28
      if (ABF16)
        ra16[t] = *reinterpret_cast<const u32x2*>(A16 + (size_t)(r0 + r) * DMODEL + k0 + c);
      else
        ra[t]   = *reinterpret_cast<const f32x4*>(A32 + (size_t)(r0 + r) * DMODEL + k0 + c);
      rw[t] = *reinterpret_cast<const f32x4*>(W + (size_t)(n0 + r) * DMODEL + k0 + c);
    }
  };

  auto store_tiles = [&](unsigned short* Asb, unsigned short* Bsb) {
#pragma unroll
    for (int t = 0; t < 4; ++t) {
      const int chunk = tid + t * 256;
      const int r = chunk >> 3;
      const int c = (chunk & 7) << 2;
      u32x2 pk;
      if (ABF16) {
        pk = ra16[t];
      } else {
        pk[0] = (unsigned int)f2bf(ra[t][0]) | ((unsigned int)f2bf(ra[t][1]) << 16);
        pk[1] = (unsigned int)f2bf(ra[t][2]) | ((unsigned int)f2bf(ra[t][3]) << 16);
      }
      *reinterpret_cast<u32x2*>(Asb + r * 40 + c) = pk;
      u32x2 wp;
      wp[0] = (unsigned int)f2bf(rw[t][0]) | ((unsigned int)f2bf(rw[t][1]) << 16);
      wp[1] = (unsigned int)f2bf(rw[t][2]) | ((unsigned int)f2bf(rw[t][3]) << 16);
      *reinterpret_cast<u32x2*>(Bsb + r * 40 + c) = wp;
    }
  };

  v8f acc[2][4];
#pragma unroll
  for (int i = 0; i < 2; ++i)
#pragma unroll
    for (int j = 0; j < 4; ++j)
#pragma unroll
      for (int e = 0; e < 8; ++e) acc[i][j][e] = 0.0f;

  // pipeline prologue
  load_tiles(0);
  store_tiles(As[0], Bs[0]);
  __syncthreads();

  for (int k0 = 0; k0 < DMODEL; k0 += 32) {
    const int  cur     = (k0 >> 5) & 1;
    const bool hasnext = (k0 + 32) < DMODEL;
    if (hasnext) load_tiles(k0 + 32);     // overlap with WMMA below

    v16bf afr[2], bfr[4];
    afr[0] = ld_frag(As[cur] + (wr * 32 +  0 + lr) * 40, half * 8);
    afr[1] = ld_frag(As[cur] + (wr * 32 + 16 + lr) * 40, half * 8);
#pragma unroll
    for (int nc = 0; nc < 4; ++nc)
      bfr[nc] = ld_frag(Bs[cur] + (wc * 64 + nc * 16 + lr) * 40, half * 8);

#pragma unroll
    for (int mr = 0; mr < 2; ++mr)
#pragma unroll
      for (int nc = 0; nc < 4; ++nc)
        acc[mr][nc] = wmma_bf16(afr[mr], bfr[nc], acc[mr][nc]);

    if (hasnext) store_tiles(As[cur ^ 1], Bs[cur ^ 1]);
    __syncthreads();
  }

  const float lnbase = 9.210340371976184f;   // ln(10000)
#pragma unroll
  for (int mr = 0; mr < 2; ++mr) {
#pragma unroll
    for (int nc = 0; nc < 4; ++nc) {
      const int n = n0 + wc * 64 + nc * 16 + lr;
#pragma unroll
      for (int r = 0; r < 8; ++r) {
        const int srow = r0 + wr * 32 + mr * 16 + r + 8 * half;
        float v = acc[mr][nc][r];
        if (ROPE) {
          // interleaved-pair RoPE; partner column n^1 lives in lane^1
          float partner = __shfl_xor(v, 1, 32);
          const int dh = n & (DHEAD - 1);
          const int jp = dh >> 1;
          const float inv = __expf(-(float)(2 * jp) * (1.0f / (float)DHEAD) * lnbase);
          const float ang = (float)srow * inv;
          float sn, cs;
          __sincosf(ang, &sn, &cs);
          v = v * cs + ((dh & 1) ? partner : -partner) * sn;
        }
        if (OUTMODE == 2) {
          reinterpret_cast<float*>(Out)[(size_t)srow * DMODEL + n] = v;
        } else if (OUTMODE == 1) {
          const int hh = n >> 7, dh2 = n & 127;
          reinterpret_cast<unsigned short*>(Out)[((size_t)hh * DHEAD + dh2) * S_LEN + srow]
              = f2bf(v);
        } else {
          const int hh = n >> 7, dh2 = n & 127;
          reinterpret_cast<unsigned short*>(Out)[((size_t)hh * S_LEN + srow) * DHEAD + dh2]
              = f2bf(v);
        }
      }
    }
  }
}

// ---------------------------------------------------------------------------
// Flash attention: block = 128 query rows x 1 head, 8 waves x 16 rows each.
// Key blocks of 32, double-buffered LDS, async global->LDS staging.
// Q/K are [H][S][DH] bf16 (RoPE'd); V is pre-transposed [H][DH][S] bf16.
// ---------------------------------------------------------------------------
__global__ __launch_bounds__(256) void flash_attn_kernel(
    const unsigned short* __restrict__ Q,
    const unsigned short* __restrict__ K,
    const unsigned short* __restrict__ Vt,  // [H][DH][S]
    unsigned short* __restrict__ O)         // [S][H*DH] bf16
{
  __shared__ unsigned short Ks[2][32 * 136];  // K block row-major, padded stride
  __shared__ unsigned short Vs[2][128 * 40];  // V^T block [dh][j], padded stride
  __shared__ unsigned short Ps[8 * 16 * 40];  // per-wave P patches [16][32]

  const int tid  = threadIdx.x;
  const int lane = tid & 31;
  const int w    = tid >> 5;
  const int lr   = lane & 15;
  const int half = lane >> 4;
  const int h    = blockIdx.y;
  const int r0   = blockIdx.x * 128;
  const int q0   = r0 + w * 16;

  const unsigned short* Qh  = Q  + (size_t)h * S_LEN * DHEAD;
  const unsigned short* Kh  = K  + (size_t)h * S_LEN * DHEAD;
  const unsigned short* VtH = Vt + (size_t)h * DHEAD * S_LEN;

  // stage one 32-key block (K row-major + V^T) into LDS buffer b
  auto stage_kv = [&](int b, int kbase) {
#pragma unroll
    for (int t = 0; t < 2; ++t) {
      const int chunk = tid + t * 256;          // 512 16-byte chunks each
      const int j  = chunk >> 4;                // key row 0..31
      const int c8 = (chunk & 15) * 8;          // dh offset 0..120
      stage16(Ks[b] + j * 136 + c8, Kh + (size_t)(kbase + j) * DHEAD + c8);
      const int dh = chunk >> 2;                // 0..127
      const int j8 = (chunk & 3) * 8;           // 0,8,16,24
      stage16(Vs[b] + dh * 40 + j8, VtH + (size_t)dh * S_LEN + kbase + j8);
    }
  };

  // Q fragments resident in registers: 4 K-steps over DH=128
  v16bf qf[4];
#pragma unroll
  for (int ks = 0; ks < 4; ++ks)
    qf[ks] = ld_frag(Qh + (size_t)(q0 + lr) * DHEAD, ks * 32 + half * 8);

  v8f o[8];
  float m[8], l[8];
#pragma unroll
  for (int t = 0; t < 8; ++t)
#pragma unroll
    for (int e = 0; e < 8; ++e) o[t][e] = 0.0f;
#pragma unroll
  for (int r = 0; r < 8; ++r) { m[r] = -1e30f; l[r] = 0.0f; }

  const float sscale = 0.08838834764831845f;  // 1/sqrt(128)
  const int nkb = (r0 >> 5) + 4;              // causal: keys up to r0+127
  unsigned short* Pw = Ps + w * 640;

  // pipeline prologue
  stage_kv(0, 0);
  wait_stage();
  __syncthreads();

  for (int kb = 0; kb < nkb; ++kb) {
    const int kbase = kb * 32;
    const int cur   = kb & 1;
    if (kb + 1 < nkb) stage_kv(cur ^ 1, kbase + 32);  // overlap with compute

    // S = Q K^T  (two 16x16 tiles over the 32 keys)
    v8f s0, s1;
#pragma unroll
    for (int e = 0; e < 8; ++e) { s0[e] = 0.0f; s1[e] = 0.0f; }
#pragma unroll
    for (int ks = 0; ks < 4; ++ks) {
      v16bf b0 = ld_frag(Ks[cur] + (      lr) * 136, ks * 32 + half * 8);
      v16bf b1 = ld_frag(Ks[cur] + (16 + lr) * 136, ks * 32 + half * 8);
      s0 = wmma_bf16(qf[ks], b0, s0);
      s1 = wmma_bf16(qf[ks], b1, s1);
    }

    // online softmax (lanes 0-15 own rows 0-7, lanes 16-31 rows 8-15)
    const bool needmask = (kbase + 31) > q0;
#pragma unroll
    for (int r = 0; r < 8; ++r) {
      const int irow = q0 + r + 8 * half;
      float a0 = s0[r] * sscale;
      float a1 = s1[r] * sscale;
      if (needmask) {
        if (kbase + lr      > irow) a0 = -1e30f;
        if (kbase + 16 + lr > irow) a1 = -1e30f;
      }
      float rm = fmaxf(a0, a1);
      rm = fmaxf(rm, __shfl_xor(rm, 1, 32));
      rm = fmaxf(rm, __shfl_xor(rm, 2, 32));
      rm = fmaxf(rm, __shfl_xor(rm, 4, 32));
      rm = fmaxf(rm, __shfl_xor(rm, 8, 32));
      const float mn = fmaxf(m[r], rm);
      const float p0 = __expf(a0 - mn);
      const float p1 = __expf(a1 - mn);
      float rs = p0 + p1;
      rs += __shfl_xor(rs, 1, 32);
      rs += __shfl_xor(rs, 2, 32);
      rs += __shfl_xor(rs, 4, 32);
      rs += __shfl_xor(rs, 8, 32);
      const float co = __expf(m[r] - mn);
      l[r] = l[r] * co + rs;
      m[r] = mn;
#pragma unroll
      for (int t = 0; t < 8; ++t) o[t][r] *= co;
      // C-layout -> A-layout transpose of P via wave-private LDS patch
      const int prow = r + 8 * half;
      Pw[prow * 40 + lr]      = f2bf(p0);
      Pw[prow * 40 + 16 + lr] = f2bf(p1);
    }

    // O += P * V
    v16bf pA = ld_frag(Pw + lr * 40, half * 8);
#pragma unroll
    for (int t = 0; t < 8; ++t) {
      v16bf vB = ld_frag(Vs[cur] + (t * 16 + lr) * 40, half * 8);
      o[t] = wmma_bf16(pA, vB, o[t]);
    }

    wait_stage();       // next buffer's async copies complete (this wave)
    __syncthreads();    // ...and everyone's, before next iteration reads/overwrites
  }

  // normalize and write bf16 [S][H*DH]
#pragma unroll
  for (int r = 0; r < 8; ++r) {
    const float inv = 1.0f / l[r];
    const int srow = q0 + r + 8 * half;
#pragma unroll
    for (int t = 0; t < 8; ++t)
      O[(size_t)srow * DMODEL + h * DHEAD + t * 16 + lr] = f2bf(o[t][r] * inv);
  }
}

// ---------------------------------------------------------------------------
extern "C" void kernel_launch(void* const* d_in, const int* in_sizes, int n_in,
                              void* d_out, int out_size, void* d_ws, size_t ws_size,
                              hipStream_t stream) {
  (void)in_sizes; (void)n_in; (void)out_size; (void)ws_size;
  const float* x  = (const float*)d_in[0];
  // d_in[1] = causal mask (implicit in kernel, unused)
  const float* wq = (const float*)d_in[2];
  const float* wk = (const float*)d_in[3];
  const float* wv = (const float*)d_in[4];
  const float* wo = (const float*)d_in[5];

  const size_t HSD = (size_t)NHEAD * S_LEN * DHEAD;   // 8,388,608 elems
  unsigned short* qb = (unsigned short*)d_ws;
  unsigned short* kb = qb + HSD;
  unsigned short* vb = kb + HSD;                      // [H][DH][S] (transposed)
  unsigned short* ab = vb + HSD;                      // [S][H*DH] bf16

  dim3 gg(DMODEL / 128, S_LEN / 128);   // 16 x 32
  dim3 bb(256);
  gemm_nt_kernel<false, true,  0><<<gg, bb, 0, stream>>>(x, wq, qb);
  gemm_nt_kernel<false, true,  0><<<gg, bb, 0, stream>>>(x, wk, kb);
  gemm_nt_kernel<false, false, 1><<<gg, bb, 0, stream>>>(x, wv, vb);

  dim3 ga(S_LEN / 128, NHEAD);          // 32 x 16
  flash_attn_kernel<<<ga, bb, 0, stream>>>(qb, kb, vb, ab);

  gemm_nt_kernel<true, false, 2><<<gg, bb, 0, stream>>>(ab, wo, (float*)d_out);
}
